// mask_Attention_49495203119784
// MI455X (gfx1250) — matrix operand order repaired and checked by reading, not compiled, not run on hardware
//
#include <hip/hip_runtime.h>
#include <hip/hip_bf16.h>
#include <math.h>

typedef __attribute__((ext_vector_type(16))) _Float16 v16h;
typedef __attribute__((ext_vector_type(8)))  _Float16 v8h;
typedef __attribute__((ext_vector_type(8)))  float    v8f;

union frag16 { v16h v; v8h h[2]; };

__device__ __forceinline__ v8f wmma16(v16h a, v16h b, v8f c) {
    return __builtin_amdgcn_wmma_f32_16x16x32_f16(false, a, false, b, (short)0, c, false, false);
}

// ---- sizes ----
#define BATCH 8
#define SEQ   1024
#define CH    768
#define H3    2304
#define NH    12
#define HD    64
#define ROWS  8192                 // BATCH*SEQ
#define QKVELTS 6291456            // BATCH*SEQ*CH (also B*H*N*HD)
#define NEGVAL (-1e9f)

__device__ __forceinline__ bool mask_ok(int br, int i, int j) {
    const bool ih = i < 9, jh = j < 9;
    const bool i0 = (i == 0), j0 = (j == 0);
    if (br == 0) return ih && jh;
    if (br == 1) return (!ih && !jh) || (i0 && !jh) || (!ih && j0) || (i0 && j0);
    return (!ih && jh) || (ih && !jh) || (i0 && j0);
}

// ---------------- f32 -> f16 convert ----------------
__global__ void cvt_f16_kernel(const float* __restrict__ in, _Float16* __restrict__ out, int n) {
    int i = blockIdx.x * blockDim.x + threadIdx.x;
    if (i < n) out[i] = (_Float16)in[i];
}

// ---------------- QKV projection: x[8192,768] @ w_qkv^T -> q,k,vT ----------------
// block = 128 threads (4 waves); wave tile = 32 rows x 64 cols (2x4 WMMA frags)
// block tile = 128 rows x 64 cols; grid (64, 36)
__global__ void qkv_gemm_kernel(const _Float16* __restrict__ xh, const _Float16* __restrict__ wh,
                                _Float16* __restrict__ qb, _Float16* __restrict__ kb,
                                _Float16* __restrict__ vtb) {
    const int lane = threadIdx.x & 31;
    const int wave = threadIdx.x >> 5;
    const int lm = lane & 15, hh = lane >> 4;
    const int rowbase = blockIdx.x * 128 + wave * 32;
    const int colbase = blockIdx.y * 64;

    const _Float16* ap0 = xh + (rowbase + lm) * CH;       // A rows (M tile 0)
    const _Float16* ap1 = ap0 + 16 * CH;                  // A rows (M tile 1)
    const _Float16* bp  = wh + (colbase + lm) * CH;       // B col n = w_qkv row

    v8f acc[2][4];
#pragma unroll
    for (int mt = 0; mt < 2; ++mt)
#pragma unroll
        for (int t = 0; t < 4; ++t) { v8f z = {}; acc[mt][t] = z; }

    for (int k = 0; k < CH; k += 32) {
        __builtin_prefetch(ap0 + k + 128, 0, 1);
        __builtin_prefetch(ap1 + k + 128, 0, 1);
        frag16 a0, a1, b[4];
        a0.h[0] = *(const v8h*)(ap0 + k + hh * 8);
        a0.h[1] = *(const v8h*)(ap0 + k + 16 + hh * 8);
        a1.h[0] = *(const v8h*)(ap1 + k + hh * 8);
        a1.h[1] = *(const v8h*)(ap1 + k + 16 + hh * 8);
#pragma unroll
        for (int t = 0; t < 4; ++t)
            { b[t].h[0] = *(const v8h*)(bp + t * 16 * CH + k + hh * 8);
              b[t].h[1] = *(const v8h*)(bp + t * 16 * CH + k + 16 + hh * 8); }
#pragma unroll
        for (int t = 0; t < 4; ++t) {
            acc[0][t] = wmma16(a0.v, b[t].v, acc[0][t]);
            acc[1][t] = wmma16(a1.v, b[t].v, acc[1][t]);
        }
    }

#pragma unroll
    for (int mt = 0; mt < 2; ++mt) {
#pragma unroll
        for (int t = 0; t < 4; ++t) {
            const int c3   = colbase + t * 16 + lm;   // column in [0, 2304)
            const int which = c3 / CH;                // 0=q 1=k 2=v (uniform per 16-col tile)
            const int c    = c3 % CH;
            const int head = c >> 6;
            const int d    = c & 63;
#pragma unroll
            for (int r = 0; r < 8; ++r) {
                const int row = rowbase + mt * 16 + r + (hh << 3);
                const int bidx = row >> 10;
                const int n    = row & 1023;
                const int bh   = bidx * NH + head;
                const float v  = acc[mt][t][r];
                if (which == 0)      qb[((bh << 10) + n) * HD + d] = (_Float16)(v * 0.125f);
                else if (which == 1) kb[((bh << 10) + n) * HD + d] = (_Float16)v;
                else                 vtb[(bh * HD + d) * SEQ + n] = (_Float16)v;
            }
        }
    }
}

// ---------------- fused tri-mask flash attention ----------------
// grid (16, 96): blockIdx.x = 64-row tile within N, blockIdx.y = b*H+h; 4 waves x 16 rows
__global__ void attn_kernel(const _Float16* __restrict__ qb, const _Float16* __restrict__ kb,
                            const _Float16* __restrict__ vtb, _Float16* __restrict__ ob) {
    __shared__ _Float16 ldsP[4][16][32];
    const int lane = threadIdx.x & 31;
    const int wave = threadIdx.x >> 5;
    const int lm = lane & 15, hh = lane >> 4;
    const int bh = blockIdx.y;
    const int rowbase = blockIdx.x * 64 + wave * 16;

    // Q A-fragments for the wave's 16 rows, two K-steps over hd=64
    const _Float16* qrow = qb + ((bh << 10) + rowbase + lm) * HD;
    frag16 qa0, qa1;
    qa0.h[0] = *(const v8h*)(qrow + hh * 8);
    qa0.h[1] = *(const v8h*)(qrow + 16 + hh * 8);
    qa1.h[0] = *(const v8h*)(qrow + 32 + hh * 8);
    qa1.h[1] = *(const v8h*)(qrow + 48 + hh * 8);

    float mrow[3][8], lrow[3][8];
    v8f O[3][4];
#pragma unroll
    for (int br = 0; br < 3; ++br) {
#pragma unroll
        for (int r = 0; r < 8; ++r) { mrow[br][r] = -1e30f; lrow[br][r] = 0.0f; }
#pragma unroll
        for (int t = 0; t < 4; ++t) { v8f z = {}; O[br][t] = z; }
    }

#pragma unroll 1
    for (int j = 0; j < SEQ; j += 32) {
        // K B-fragments: keys j..j+15 and j+16..j+31, two hd K-steps each
        const _Float16* krow0 = kb + ((bh << 10) + j + lm) * HD;
        const _Float16* krow1 = krow0 + 16 * HD;
        frag16 kb00, kb01, kb10, kb11;
        kb00.h[0] = *(const v8h*)(krow0 + hh * 8);
        kb00.h[1] = *(const v8h*)(krow0 + 16 + hh * 8);
        kb01.h[0] = *(const v8h*)(krow0 + 32 + hh * 8);
        kb01.h[1] = *(const v8h*)(krow0 + 48 + hh * 8);
        kb10.h[0] = *(const v8h*)(krow1 + hh * 8);
        kb10.h[1] = *(const v8h*)(krow1 + 16 + hh * 8);
        kb11.h[0] = *(const v8h*)(krow1 + 32 + hh * 8);
        kb11.h[1] = *(const v8h*)(krow1 + 48 + hh * 8);

        v8f S0 = {}, S1 = {};
        S0 = wmma16(qa0.v, kb00.v, S0);
        S0 = wmma16(qa1.v, kb01.v, S0);
        S1 = wmma16(qa0.v, kb10.v, S1);
        S1 = wmma16(qa1.v, kb11.v, S1);

        // V B-fragments for P(16x32) @ V(32x64): 4 tiles of 16 output dims
        frag16 vb[4];
#pragma unroll
        for (int t = 0; t < 4; ++t)
            { vb[t].h[0] = *(const v8h*)(vtb + (bh * HD + t * 16 + lm) * SEQ + j + hh * 8);
              vb[t].h[1] = *(const v8h*)(vtb + (bh * HD + t * 16 + lm) * SEQ + j + 16 + hh * 8); }

        const int j0 = j + lm, j1 = j0 + 16;

        for (int br = 0; br < 3; ++br) {
            float p0[8], p1[8];
#pragma unroll
            for (int r = 0; r < 8; ++r) {
                const int i = rowbase + r + (hh << 3);
                float s0 = mask_ok(br, i, j0) ? S0[r] : NEGVAL;
                float s1 = mask_ok(br, i, j1) ? S1[r] : NEGVAL;
                float tm = fmaxf(s0, s1);
                tm = fmaxf(tm, __shfl_xor(tm, 1, 32));
                tm = fmaxf(tm, __shfl_xor(tm, 2, 32));
                tm = fmaxf(tm, __shfl_xor(tm, 4, 32));
                tm = fmaxf(tm, __shfl_xor(tm, 8, 32));
                const float mn = fmaxf(mrow[br][r], tm);
                const float corr = __expf(mrow[br][r] - mn);
                const float e0 = __expf(s0 - mn);
                const float e1 = __expf(s1 - mn);
                float rs = e0 + e1;
                rs += __shfl_xor(rs, 1, 32);
                rs += __shfl_xor(rs, 2, 32);
                rs += __shfl_xor(rs, 4, 32);
                rs += __shfl_xor(rs, 8, 32);
                lrow[br][r] = lrow[br][r] * corr + rs;
                mrow[br][r] = mn;
#pragma unroll
                for (int t = 0; t < 4; ++t) O[br][t][r] *= corr;
                p0[r] = e0; p1[r] = e1;
            }
            // stage P through LDS: C/D layout -> row-major 16x32 -> A-fragment
#pragma unroll
            for (int r = 0; r < 8; ++r) {
                ldsP[wave][r + (hh << 3)][lm]      = (_Float16)p0[r];
                ldsP[wave][r + (hh << 3)][lm + 16] = (_Float16)p1[r];
            }
            frag16 pa;
            pa.h[0] = *(const v8h*)(&ldsP[wave][lm][hh * 8]);
            pa.h[1] = *(const v8h*)(&ldsP[wave][lm][16 + hh * 8]);
#pragma unroll
            for (int t = 0; t < 4; ++t) O[br][t] = wmma16(pa.v, vb[t].v, O[br][t]);
        }
    }

    // epilogue: O / l -> f16 context buffer [3][B,N,C]
    const int b = bh / NH, head = bh % NH;
#pragma unroll
    for (int br = 0; br < 3; ++br) {
#pragma unroll
        for (int r = 0; r < 8; ++r) {
            const float inv = 1.0f / lrow[br][r];
            const int row = rowbase + r + (hh << 3);
            _Float16* dst = ob + (size_t)br * QKVELTS + ((b << 10) + row) * CH + (head << 6);
#pragma unroll
            for (int t = 0; t < 4; ++t) dst[t * 16 + lm] = (_Float16)(O[br][t][r] * inv);
        }
    }
}

// ---------------- output projection: ob[br] @ w_proj^T + b_proj -> f32 ----------------
// grid (64, 12, 3); block 128 threads = 4 waves; wave tile = 32 rows x 64 cols
__global__ void proj_gemm_kernel(const _Float16* __restrict__ ob, const _Float16* __restrict__ wph,
                                 const float* __restrict__ bproj, float* __restrict__ out) {
    const int br = blockIdx.z;
    const int lane = threadIdx.x & 31;
    const int wave = threadIdx.x >> 5;
    const int lm = lane & 15, hh = lane >> 4;
    const int rowbase = blockIdx.x * 128 + wave * 32;
    const int colbase = blockIdx.y * 64;

    const _Float16* ap0 = ob + (size_t)br * QKVELTS + (rowbase + lm) * CH;
    const _Float16* ap1 = ap0 + 16 * CH;
    const _Float16* bp  = wph + (colbase + lm) * CH;

    v8f acc[2][4];
#pragma unroll
    for (int mt = 0; mt < 2; ++mt)
#pragma unroll
        for (int t = 0; t < 4; ++t) { v8f z = {}; acc[mt][t] = z; }

    for (int k = 0; k < CH; k += 32) {
        __builtin_prefetch(ap0 + k + 128, 0, 1);
        __builtin_prefetch(ap1 + k + 128, 0, 1);
        frag16 a0, a1, b[4];
        a0.h[0] = *(const v8h*)(ap0 + k + hh * 8);
        a0.h[1] = *(const v8h*)(ap0 + k + 16 + hh * 8);
        a1.h[0] = *(const v8h*)(ap1 + k + hh * 8);
        a1.h[1] = *(const v8h*)(ap1 + k + 16 + hh * 8);
#pragma unroll
        for (int t = 0; t < 4; ++t)
            { b[t].h[0] = *(const v8h*)(bp + t * 16 * CH + k + hh * 8);
              b[t].h[1] = *(const v8h*)(bp + t * 16 * CH + k + 16 + hh * 8); }
#pragma unroll
        for (int t = 0; t < 4; ++t) {
            acc[0][t] = wmma16(a0.v, b[t].v, acc[0][t]);
            acc[1][t] = wmma16(a1.v, b[t].v, acc[1][t]);
        }
    }

    float* obase = out + (size_t)br * QKVELTS;
#pragma unroll
    for (int mt = 0; mt < 2; ++mt) {
#pragma unroll
        for (int t = 0; t < 4; ++t) {
            const int col = colbase + t * 16 + lm;
            const float bias = bproj[col];
#pragma unroll
            for (int r = 0; r < 8; ++r) {
                const int row = rowbase + mt * 16 + r + (hh << 3);
                obase[(size_t)row * CH + col] = acc[mt][t][r] + bias;
            }
        }
    }
}

extern "C" void kernel_launch(void* const* d_in, const int* in_sizes, int n_in,
                              void* d_out, int out_size, void* d_ws, size_t ws_size,
                              hipStream_t stream) {
    const float* x      = (const float*)d_in[0];   // [8,1024,768]
    const float* w_qkv  = (const float*)d_in[1];   // [2304,768]
    const float* w_proj = (const float*)d_in[2];   // [768,768]
    const float* b_proj = (const float*)d_in[3];   // [768]
    float* out = (float*)d_out;                    // 3 x [8,1024,768] f32

    _Float16* ws    = (_Float16*)d_ws;
    _Float16* xh    = ws;                          // 6291456
    _Float16* wqkvh = xh    + QKVELTS;             // 1769472
    _Float16* wprojh= wqkvh + (size_t)H3 * CH;     // 589824
    _Float16* qb    = wprojh+ (size_t)CH * CH;     // 6291456
    _Float16* kb    = qb    + QKVELTS;             // 6291456
    _Float16* vtb   = kb    + QKVELTS;             // 6291456
    _Float16* ob    = vtb   + QKVELTS;             // 3 * 6291456

    cvt_f16_kernel<<<(QKVELTS + 255) / 256, 256, 0, stream>>>(x, xh, QKVELTS);
    cvt_f16_kernel<<<(H3 * CH + 255) / 256, 256, 0, stream>>>(w_qkv, wqkvh, H3 * CH);
    cvt_f16_kernel<<<(CH * CH + 255) / 256, 256, 0, stream>>>(w_proj, wprojh, CH * CH);

    qkv_gemm_kernel<<<dim3(ROWS / 128, H3 / 64), 128, 0, stream>>>(xh, wqkvh, qb, kb, vtb);
    attn_kernel<<<dim3(SEQ / 64, BATCH * NH), 128, 0, stream>>>(qb, kb, vtb, ob);
    proj_gemm_kernel<<<dim3(ROWS / 128, CH / 64, 3), 128, 0, stream>>>(ob, wprojh, b_proj, out);
}
